// PretrainedMoE_18949395710016
// MI455X (gfx1250) — compile-verified, matrix operand order checked by reading, not run on hardware
//
#include <hip/hip_runtime.h>
#include <hip/hip_bf16.h>
#include <math.h>

// ---------------------------------------------------------------------------
// MoE forward for MI455X (gfx1250), wave32 + WMMA.
//   B=4096 rows, D=768, E=16 experts, H=768, C=1000 classes, TOPK=4.
// Outputs (concatenated f32 in d_out):
//   market_probs [B,C] | all_probs [E,B,C] | gate_probs [B,E]
// ---------------------------------------------------------------------------

#define BB   4096
#define DD   768
#define EE   16
#define HH   768
#define CC   1000
#define KTOP 4

#define KT1  (DD / 32)       // 24 K-tiles for GEMM1
#define NT1  (HH / 16)       // 48 N-tiles for GEMM1
#define KT2  (HH / 32)       // 24 K-tiles for GEMM2
#define NT2  63              // ceil(1000/16) N-tiles for GEMM2 (padded)
#define BM   64              // rows per block in the fused expert kernel
#define YPITCH (DD + 8)      // LDS row pitch in halves (16B-aligned rows)
#define LDS_BYTES ((size_t)BM * YPITCH * sizeof(_Float16))   // 99,328 B (<320KB/WGP)

typedef __attribute__((ext_vector_type(16))) _Float16 v16h;
typedef __attribute__((ext_vector_type(8)))  float    v8f;

struct alignas(16) f4 { float x, y, z, w; };
struct f4x2 { f4 lo, hi; };

// ---------------------------------------------------------------------------
// Pack w1 (f32 [E,D,H]) -> f16 WMMA-B-fragment layout:
//   w1p[e][kt][nt][lane][j] : 32x16 K-major tile, 16 contiguous halves/lane.
//   lane = hb*16 + n  (hb = K half-block, n = column 0..15)
//   j    = (klocal&7) + 8*(klocal>=16)
// One thread per (e,kt,nt,klocal): reads 16 contiguous f32, writes 16 halves.
// ---------------------------------------------------------------------------
__global__ void moe_pack_w1(const float* __restrict__ w1,
                            _Float16* __restrict__ w1p) {
    size_t tid = (size_t)blockIdx.x * blockDim.x + threadIdx.x;
    const size_t total = (size_t)EE * KT1 * NT1 * 32;
    if (tid >= total) return;
    int klocal = tid & 31;
    size_t tile = tid >> 5;
    int nt = (int)(tile % NT1);
    int kt = (int)((tile / NT1) % KT1);
    int e  = (int)(tile / ((size_t)NT1 * KT1));
    int k  = kt * 32 + klocal;
    int j  = (klocal & 7) + ((klocal >> 4) << 3);
    int hb = (klocal >> 3) & 1;
    const float* src = w1 + ((size_t)e * DD + k) * HH + nt * 16;
    _Float16* dst = w1p + tile * 512 + hb * 256 + j;
#pragma unroll
    for (int n = 0; n < 16; ++n) dst[n * 16] = (_Float16)src[n];
}

// Same for w2 (f32 [E,H,C]) with N padded to NT2*16 (zero fill past C).
__global__ void moe_pack_w2(const float* __restrict__ w2,
                            _Float16* __restrict__ w2p) {
    size_t tid = (size_t)blockIdx.x * blockDim.x + threadIdx.x;
    const size_t total = (size_t)EE * KT2 * NT2 * 32;
    if (tid >= total) return;
    int klocal = tid & 31;
    size_t tile = tid >> 5;
    int nt = (int)(tile % NT2);
    int kt = (int)((tile / NT2) % KT2);
    int e  = (int)(tile / ((size_t)NT2 * KT2));
    int k  = kt * 32 + klocal;
    int j  = (klocal & 7) + ((klocal >> 4) << 3);
    int hb = (klocal >> 3) & 1;
    const float* src = w2 + ((size_t)e * HH + k) * CC;
    _Float16* dst = w2p + tile * 512 + hb * 256 + j;
#pragma unroll
    for (int n = 0; n < 16; ++n) {
        int col = nt * 16 + n;
        float v = (col < CC) ? src[col] : 0.0f;
        dst[n * 16] = (_Float16)v;
    }
}

// ---------------------------------------------------------------------------
// Router + LayerNorm stats. One wave32 per row (4 waves / 128-thread block).
// Writes: gate_probs (d_out tail), topk idx+weights (ws), mu/rstd (ws).
// ---------------------------------------------------------------------------
__global__ void moe_router(const float* __restrict__ features,
                           const float* __restrict__ router_w,
                           const float* __restrict__ router_b,
                           float* __restrict__ gate_out,   // [B,E]
                           int*   __restrict__ tkidx,      // [B,4]
                           float* __restrict__ tkw,        // [B,4]
                           float* __restrict__ stats) {    // [B,2] mu,rstd
    const int lane = threadIdx.x & 31;
    const int wv   = threadIdx.x >> 5;
    const int row  = blockIdx.x * 4 + wv;
    if (row >= BB) return;

    float acc[EE];
#pragma unroll
    for (int e = 0; e < EE; ++e) acc[e] = 0.0f;
    float s = 0.0f, s2 = 0.0f;

    const float* frow = features + (size_t)row * DD;
    for (int d = lane; d < DD; d += 32) {
        float f = frow[d];
        s += f; s2 += f * f;
        const float* rp = router_w + (size_t)d * EE;
#pragma unroll
        for (int e = 0; e < EE; ++e) acc[e] += f * rp[e];
    }
    // wave32 xor-reduction
#pragma unroll
    for (int o = 16; o; o >>= 1) {
        s  += __shfl_xor(s, o, 32);
        s2 += __shfl_xor(s2, o, 32);
#pragma unroll
        for (int e = 0; e < EE; ++e) acc[e] += __shfl_xor(acc[e], o, 32);
    }

    if (lane == 0) {
        float logit[EE], p[EE];
        float mx = -3.0e38f;
#pragma unroll
        for (int e = 0; e < EE; ++e) { logit[e] = acc[e] + router_b[e]; mx = fmaxf(mx, logit[e]); }
        float sum = 0.0f;
#pragma unroll
        for (int e = 0; e < EE; ++e) { p[e] = __expf(logit[e] - mx); sum += p[e]; }
        float inv = 1.0f / sum;
#pragma unroll
        for (int e = 0; e < EE; ++e) { p[e] *= inv; gate_out[(size_t)row * EE + e] = p[e]; }

        // top-4 (stable argmax, matches lax.top_k first-occurrence ties)
        int   bi[KTOP];
        float bv[KTOP];
        bool used[EE];
#pragma unroll
        for (int e = 0; e < EE; ++e) used[e] = false;
        float tsum = 0.0f;
#pragma unroll
        for (int k = 0; k < KTOP; ++k) {
            int best = 0; float bvv = -1.0f;
#pragma unroll
            for (int e = 0; e < EE; ++e)
                if (!used[e] && p[e] > bvv) { bvv = p[e]; best = e; }
            used[best] = true; bi[k] = best; bv[k] = bvv; tsum += bvv;
        }
        float tinv = 1.0f / tsum;
#pragma unroll
        for (int k = 0; k < KTOP; ++k) {
            tkidx[row * KTOP + k] = bi[k];
            tkw[row * KTOP + k]   = bv[k] * tinv;
        }
        float mu  = s * (1.0f / DD);
        float var = s2 * (1.0f / DD) - mu * mu;
        stats[row * 2 + 0] = mu;
        stats[row * 2 + 1] = rsqrtf(var + 1e-5f);
    }
}

// ---------------------------------------------------------------------------
// Fused expert kernel: block = (64 rows of B) x (1 expert), 256 thr = 8 waves.
//   Dynamic LDS tile (99KB, CDNA5 WGP has 320KB):
//     y = (xn*ln_g+ln_b) f16 [64 x 768] -> GEMM1 (WMMA f16, 24 tiles/wave)
//     -> GELU -> h reuses the same LDS buffer
//     -> GEMM2 (4 row-tiles per w2 fragment: 64 FLOP per L2 byte)
//     -> raw logits written to the all_probs slot of d_out.
// ---------------------------------------------------------------------------
__global__ void __launch_bounds__(256)
moe_expert(const float* __restrict__ features,
           const float* __restrict__ ln_g, const float* __restrict__ ln_b,
           const _Float16* __restrict__ w1p, const float* __restrict__ b1,
           const _Float16* __restrict__ w2p, const float* __restrict__ b2,
           const float* __restrict__ stats,
           float* __restrict__ all_logits /* [E,B,C] slot of d_out */) {
    extern __shared__ _Float16 ysh[];   // [BM][YPITCH]

    const int e    = blockIdx.y;
    const int row0 = blockIdx.x * BM;
    const int tid  = threadIdx.x;
    const int lane = tid & 31;
    const int wv   = tid >> 5;     // 0..7
    const int hb   = lane >> 4;    // K half-block
    const int ln   = lane & 15;    // column / row-in-half

    // ---- Stage A: build y = (x-mu)*rstd*g + b as f16 in LDS --------------
    const float* lng = ln_g + (size_t)e * DD;
    const float* lnb = ln_b + (size_t)e * DD;
    for (int r = 0; r < BM; ++r) {
        const float mu  = stats[(row0 + r) * 2 + 0];
        const float rsd = stats[(row0 + r) * 2 + 1];
        const float* frow = features + (size_t)(row0 + r) * DD;
#pragma unroll
        for (int cc = 0; cc < 3; ++cc) {
            int c = cc * 256 + tid;
            float yv = (frow[c] - mu) * rsd * lng[c] + lnb[c];
            ysh[r * YPITCH + c] = (_Float16)yv;
        }
    }
    __syncthreads();

    // ---- GEMM1: [64x768] x [768x768] + b1, GELU --------------------------
    // wave -> row-tile rt = wv>>1 (4 x 16 rows), 24-N-tile strip = wv&1.
    const int rt    = wv >> 1;
    const int strip = wv & 1;
    const _Float16* w1e = w1p + (size_t)e * KT1 * NT1 * 512;
    const float*    b1e = b1 + (size_t)e * HH;

    v8f acc[24];
#pragma unroll
    for (int j = 0; j < 24; ++j) {
        float bv = b1e[(strip * 24 + j) * 16 + ln];
#pragma unroll
        for (int r = 0; r < 8; ++r) acc[j][r] = bv;
    }

    for (int kt = 0; kt < KT1; ++kt) {
        const int k0 = kt * 32;
        const int arow = (rt * 16 + ln) * YPITCH;
        f4x2 at;
        at.lo = *(const f4*)&ysh[arow + k0 + hb * 8];
        at.hi = *(const f4*)&ysh[arow + k0 + 16 + hb * 8];
        v16h a = __builtin_bit_cast(v16h, at);
        if (kt + 1 < KT1)
            __builtin_prefetch(w1e + (((size_t)(kt + 1) * NT1 + strip * 24) * 32 + lane) * 16, 0, 3);
#pragma unroll
        for (int j = 0; j < 24; ++j) {
            const int nt = strip * 24 + j;
            v16h bfrag = *(const v16h*)(w1e + (((size_t)kt * NT1 + nt) * 32 + lane) * 16);
            acc[j] = __builtin_amdgcn_wmma_f32_16x16x32_f16(
                false, a, false, bfrag, (short)0, acc[j], false, false);
        }
    }

    __syncthreads();           // everyone done reading y tile
    // GELU epilogue -> h (f16) overwrites the same LDS buffer
#pragma unroll
    for (int j = 0; j < 24; ++j) {
        const int nt = strip * 24 + j;
#pragma unroll
        for (int r = 0; r < 8; ++r) {
            float x = acc[j][r];
            float g = 0.5f * x * (1.0f + erff(x * 0.70710678f));
            ysh[(rt * 16 + r + hb * 8) * YPITCH + nt * 16 + ln] = (_Float16)g;
        }
    }
    __syncthreads();

    // ---- GEMM2: [64x768] x [768x1008] + b2 -> logits ---------------------
    // 4 row-tiles share each w2 fragment (64 FLOP per fragment byte).
    const _Float16* w2e = w2p + (size_t)e * KT2 * NT2 * 512;
    const float*    b2e = b2 + (size_t)e * CC;
    float* outbase = all_logits + ((size_t)e * BB + row0) * CC;

    for (int ct = wv; ct < NT2; ct += 8) {
        const int col = ct * 16 + ln;
        float bv = (col < CC) ? b2e[col] : 0.0f;
        v8f c4[4];
#pragma unroll
        for (int q = 0; q < 4; ++q)
#pragma unroll
            for (int r = 0; r < 8; ++r) c4[q][r] = bv;

        for (int kt = 0; kt < KT2; ++kt) {
            const int k0 = kt * 32;
            v16h bfrag = *(const v16h*)(w2e + (((size_t)kt * NT2 + ct) * 32 + lane) * 16);
            if (kt + 1 < KT2)
                __builtin_prefetch(w2e + (((size_t)(kt + 1) * NT2 + ct) * 32 + lane) * 16, 0, 3);
#pragma unroll
            for (int q = 0; q < 4; ++q) {
                const int arow = (q * 16 + ln) * YPITCH;
                f4x2 at;
                at.lo = *(const f4*)&ysh[arow + k0 + hb * 8];
                at.hi = *(const f4*)&ysh[arow + k0 + 16 + hb * 8];
                v16h a = __builtin_bit_cast(v16h, at);
                c4[q] = __builtin_amdgcn_wmma_f32_16x16x32_f16(
                    false, a, false, bfrag, (short)0, c4[q], false, false);
            }
        }
        if (col < CC) {
            float* dst = outbase + col;
#pragma unroll
            for (int q = 0; q < 4; ++q)
#pragma unroll
                for (int r = 0; r < 8; ++r)
                    dst[(size_t)(q * 16 + r + hb * 8) * CC] = c4[q][r];
        }
    }
}

// ---------------------------------------------------------------------------
// Softmax over classes (in place on all_probs) + top-4 weighted mixture.
// One block per row b; 256 threads cover 1000 classes (4 each).
// ---------------------------------------------------------------------------
__device__ inline float blk_reduce(float v, float* red, bool is_max) {
    const int lane = threadIdx.x & 31, wid = threadIdx.x >> 5;
#pragma unroll
    for (int o = 16; o; o >>= 1) {
        float t = __shfl_xor(v, o, 32);
        v = is_max ? fmaxf(v, t) : v + t;
    }
    if (lane == 0) red[wid] = v;
    __syncthreads();
    if (wid == 0) {
        float t = (lane < 8) ? red[lane] : (is_max ? -3.0e38f : 0.0f);
#pragma unroll
        for (int o = 4; o; o >>= 1) {
            float u = __shfl_xor(t, o, 32);
            t = is_max ? fmaxf(t, u) : t + u;
        }
        if (lane == 0) red[0] = t;
    }
    __syncthreads();
    float r = red[0];
    __syncthreads();
    return r;
}

__global__ void moe_softmax_mix(float* __restrict__ out,
                                const int* __restrict__ tkidx,
                                const float* __restrict__ tkw) {
    __shared__ float red[8];
    const int b = blockIdx.x, tid = threadIdx.x;
    float* allp   = out + (size_t)BB * CC;
    float* market = out + (size_t)b * CC;

    int   ti[KTOP];
    float twv[KTOP];
#pragma unroll
    for (int k = 0; k < KTOP; ++k) { ti[k] = tkidx[b * KTOP + k]; twv[k] = tkw[b * KTOP + k]; }

    float mix[4] = {0.f, 0.f, 0.f, 0.f};
    for (int e = 0; e < EE; ++e) {
        float* row = allp + ((size_t)e * BB + b) * CC;
        float x[4];
#pragma unroll
        for (int q = 0; q < 4; ++q) {
            int c = tid + q * 256;
            x[q] = (c < CC) ? row[c] : -3.0e38f;
        }
        float mx = fmaxf(fmaxf(x[0], x[1]), fmaxf(x[2], x[3]));
        mx = blk_reduce(mx, red, true);
        float ex[4], s = 0.f;
#pragma unroll
        for (int q = 0; q < 4; ++q) { ex[q] = __expf(x[q] - mx); s += ex[q]; }
        s = blk_reduce(s, red, false);
        float inv = 1.0f / s;
        float we = 0.f;
#pragma unroll
        for (int k = 0; k < KTOP; ++k) we = (e == ti[k]) ? twv[k] : we;
#pragma unroll
        for (int q = 0; q < 4; ++q) {
            int c = tid + q * 256;
            if (c < CC) {
                float p = ex[q] * inv;
                row[c] = p;
                mix[q] += we * p;
            }
        }
    }
#pragma unroll
    for (int q = 0; q < 4; ++q) {
        int c = tid + q * 256;
        if (c < CC) market[c] = mix[q];
    }
}

// ---------------------------------------------------------------------------
extern "C" void kernel_launch(void* const* d_in, const int* in_sizes, int n_in,
                              void* d_out, int out_size, void* d_ws, size_t ws_size,
                              hipStream_t stream) {
    const float* features = (const float*)d_in[0];
    const float* router_w = (const float*)d_in[1];
    const float* router_b = (const float*)d_in[2];
    const float* ln_g     = (const float*)d_in[3];
    const float* ln_b     = (const float*)d_in[4];
    const float* w1       = (const float*)d_in[5];
    const float* b1       = (const float*)d_in[6];
    const float* w2       = (const float*)d_in[7];
    const float* b2       = (const float*)d_in[8];
    float* out = (float*)d_out;

    // Workspace layout (~43.9 MB)
    char* ws = (char*)d_ws;
    const size_t sz_w1p = (size_t)EE * KT1 * NT1 * 512 * sizeof(_Float16); // 18.9 MB
    const size_t sz_w2p = (size_t)EE * KT2 * NT2 * 512 * sizeof(_Float16); // 24.8 MB
    _Float16* w1p  = (_Float16*)ws;
    _Float16* w2p  = (_Float16*)(ws + sz_w1p);
    float*    stats = (float*)(ws + sz_w1p + sz_w2p);                   // B*2 f32
    int*      tkidx = (int*)(ws + sz_w1p + sz_w2p + (size_t)BB * 2 * 4);
    float*    tkw   = (float*)((char*)tkidx + (size_t)BB * KTOP * 4);

    // d_out regions
    float* all_logits = out + (size_t)BB * CC;                 // [E,B,C]
    float* gate_out   = out + (size_t)BB * CC + (size_t)EE * BB * CC; // [B,E]

    // 1) weight pack (f32 -> f16, WMMA fragment layout)
    {
        size_t n1 = (size_t)EE * KT1 * NT1 * 32;
        size_t n2 = (size_t)EE * KT2 * NT2 * 32;
        moe_pack_w1<<<(unsigned)((n1 + 255) / 256), 256, 0, stream>>>(w1, w1p);
        moe_pack_w2<<<(unsigned)((n2 + 255) / 256), 256, 0, stream>>>(w2, w2p);
    }
    // 2) router + LN stats
    moe_router<<<BB / 4, 128, 0, stream>>>(features, router_w, router_b,
                                           gate_out, tkidx, tkw, stats);
    // 3) fused per-expert WMMA pipeline (99KB dynamic LDS per block)
    {
        dim3 grid(BB / BM, EE);
        moe_expert<<<grid, 256, LDS_BYTES, stream>>>(features, ln_g, ln_b,
                                                     w1p, b1, w2p, b2, stats, all_logits);
    }
    // 4) softmax + top-4 mixture
    moe_softmax_mix<<<BB, 256, 0, stream>>>(out, tkidx, tkw);
}
